// NodeGAT_3985729651397
// MI455X (gfx1250) — compile-verified, matrix operand order checked by reading, not run on hardware
//
#include <hip/hip_runtime.h>
#include <hip/hip_fp16.h>
#include <math.h>

typedef __attribute__((ext_vector_type(16))) _Float16 v16h;
typedef __attribute__((ext_vector_type(8)))  _Float16 v8h;
typedef __attribute__((ext_vector_type(8)))  float    v8f;

#define N_NODES 50000
#define N_EDGES 800000
#define ET      (N_EDGES + N_NODES)   // edges + self loops
#define K_DIM   128                   // input width of every layer
#define KSTEPS  (K_DIM / 32)          // 4 wmma k-steps

// -------------------------------------------------------------------------
// elementwise helpers
// -------------------------------------------------------------------------
__global__ void f32_to_f16_kernel(const float* __restrict__ in,
                                  _Float16* __restrict__ out, int n) {
    int i = blockIdx.x * blockDim.x + threadIdx.x;
    if (i < n) out[i] = (_Float16)in[i];
}

__global__ void fill_f32_kernel(float* __restrict__ p, float v, int n) {
    int i = blockIdx.x * blockDim.x + threadIdx.x;
    if (i < n) p[i] = v;
}

__device__ inline void atomicMaxF(float* addr, float val) {
    if (!(val == val)) return;  // skip NaN
    if (val >= 0.0f) atomicMax((int*)addr, __float_as_int(val));
    else             atomicMin((unsigned int*)addr, __float_as_uint(val));
}

// -------------------------------------------------------------------------
// Pack W[128,N] (f32 row-major) into the exact B-operand lane layout for
// v_wmma_f32_16x16x32_f16, fused with f32->f16 conversion and zero padding.
// Layout: Wp[tn][kstep][lane][j] with
//   half = lane>>4, sub = lane&15, col = tn*16+sub
//   k    = kstep*32 + (j<8 ? half*8+j : 16+half*8+(j-8))
// GEMM then reads B as two contiguous 128-bit loads per lane.
// -------------------------------------------------------------------------
__global__ void pack_w_kernel(const float* __restrict__ W,
                              _Float16* __restrict__ Wp,
                              int N, int tiles_n) {
    int idx = blockIdx.x * blockDim.x + threadIdx.x;
    int total = tiles_n * KSTEPS * 32 * 16;
    if (idx >= total) return;
    int j    = idx & 15;
    int lane = (idx >> 4) & 31;
    int ks   = (idx >> 9) & 3;
    int tn   = idx >> 11;
    int half = lane >> 4;
    int sub  = lane & 15;
    int k    = ks * 32 + (j < 8 ? half * 8 + j : 16 + half * 8 + (j - 8));
    int col  = tn * 16 + sub;
    Wp[idx]  = (col < N) ? (_Float16)W[(size_t)k * N + col] : (_Float16)0.0f;
}

// -------------------------------------------------------------------------
// Fused dual WMMA GEMM: Yl = X @ Wl, Yr = X @ Wr
//   X  : [M,128] f16 row-major (A operand, loaded once)
//   Wp*: packed B operands (see pack_w_kernel)
// one wave32 per 16x16 output tile position, 8 wmma ops per wave
// -------------------------------------------------------------------------
__global__ void gemm2_wmma_f16_kernel(const _Float16* __restrict__ X,
                                      const _Float16* __restrict__ Wpl,
                                      const _Float16* __restrict__ Wpr,
                                      float* __restrict__ Yl,
                                      float* __restrict__ Yr,
                                      int M, int N, int tiles_n) {
    const int wid  = blockIdx.x * (blockDim.x >> 5) + (threadIdx.x >> 5);
    const int lane = threadIdx.x & 31;
    const int tiles_m = M >> 4;                 // M is a multiple of 16
    if (wid >= tiles_m * tiles_n) return;       // wave-uniform exit
    const int tm = wid / tiles_n;
    const int tn = wid % tiles_n;

    const int half = lane >> 4;                 // 0 | 1
    const int sub  = lane & 15;                 // A-row / B-col within tile

    const _Float16* arowp = X + ((size_t)(tm * 16 + sub)) * K_DIM;
    const _Float16* blp   = Wpl + ((size_t)tn * KSTEPS * 32 + lane) * 16;
    const _Float16* brp   = Wpr + ((size_t)tn * KSTEPS * 32 + lane) * 16;

    v8f accl = {};
    v8f accr = {};
    #pragma unroll
    for (int ks = 0; ks < KSTEPS; ++ks) {
        const int k0 = ks * 32;
        if (ks + 1 < KSTEPS)
            __builtin_prefetch(arowp + k0 + 32, 0, 0);   // global_prefetch_b8

        // A tile 16x32: lanes 0-15 row M=sub with K 0-7|16-23, lanes 16-31 K 8-15|24-31
        union { v16h v; v8h h[2]; } A;
        A.h[0] = *(const v8h*)(arowp + k0 + half * 8);
        A.h[1] = *(const v8h*)(arowp + k0 + 16 + half * 8);

        // B tiles: pre-packed, two contiguous 128-bit loads each
        union { v16h v; v8h h[2]; } Bl, Br;
        Bl.h[0] = *(const v8h*)(blp + (size_t)ks * 32 * 16);
        Bl.h[1] = *(const v8h*)(blp + (size_t)ks * 32 * 16 + 8);
        Br.h[0] = *(const v8h*)(brp + (size_t)ks * 32 * 16);
        Br.h[1] = *(const v8h*)(brp + (size_t)ks * 32 * 16 + 8);

        accl = __builtin_amdgcn_wmma_f32_16x16x32_f16(
                   false, A.v, false, Bl.v, (short)0, accl, false, false);
        accr = __builtin_amdgcn_wmma_f32_16x16x32_f16(
                   false, A.v, false, Br.v, (short)0, accr, false, false);
    }

    // C/D layout: VGPR r -> row = tm*16 + half*8 + r, col = tn*16 + sub
    const int bcol = tn * 16 + sub;
    if (bcol < N) {
        const int row0 = tm * 16 + half * 8;
        #pragma unroll
        for (int r = 0; r < 8; ++r) {
            Yl[(size_t)(row0 + r) * N + bcol] = accl[r];
            Yr[(size_t)(row0 + r) * N + bcol] = accr[r];
        }
    }
}

// -------------------------------------------------------------------------
// per-edge attention logit + segment max   (one thread per (edge, head))
// -------------------------------------------------------------------------
__global__ void edge_logit_kernel(const float* __restrict__ xl,
                                  const float* __restrict__ xr,
                                  const int* __restrict__ src,
                                  const int* __restrict__ dst,
                                  const float* __restrict__ att,
                                  float* __restrict__ logit,
                                  float* __restrict__ mbuf,
                                  int H, int C) {
    int tid = blockIdx.x * blockDim.x + threadIdx.x;
    if (tid >= ET * H) return;
    int e = tid / H, h = tid - e * H;
    int s, d;
    if (e < N_EDGES) { s = src[e]; d = dst[e]; } else { s = d = e - N_EDGES; }
    const int HC = H * C;
    const float4* pl = (const float4*)(xl + (size_t)s * HC + h * C);
    const float4* pr = (const float4*)(xr + (size_t)d * HC + h * C);
    const float4* pa = (const float4*)(att + h * C);
    float acc = 0.0f;
    for (int c4 = 0; c4 < (C >> 2); ++c4) {
        float4 a = pl[c4], b = pr[c4], w = pa[c4];
        float v;
        v = a.x + b.x; v = v > 0.0f ? v : 0.2f * v; acc += w.x * v;
        v = a.y + b.y; v = v > 0.0f ? v : 0.2f * v; acc += w.y * v;
        v = a.z + b.z; v = v > 0.0f ? v : 0.2f * v; acc += w.z * v;
        v = a.w + b.w; v = v > 0.0f ? v : 0.2f * v; acc += w.w * v;
    }
    logit[tid] = acc;
    atomicMaxF(mbuf + (size_t)d * H + h, acc);
}

// -------------------------------------------------------------------------
// exp(logit - max) in place + segment sum of denominators
// -------------------------------------------------------------------------
__global__ void edge_expsum_kernel(const int* __restrict__ dst,
                                   const float* __restrict__ mbuf,
                                   float* __restrict__ logit,
                                   float* __restrict__ den, int H) {
    int tid = blockIdx.x * blockDim.x + threadIdx.x;
    if (tid >= ET * H) return;
    int e = tid / H, h = tid - e * H;
    int d = (e < N_EDGES) ? dst[e] : e - N_EDGES;
    float ex = expf(logit[tid] - mbuf[(size_t)d * H + h]);
    logit[tid] = ex;
    atomicAdd(den + (size_t)d * H + h, ex);
}

// -------------------------------------------------------------------------
// weighted scatter-add aggregation (one thread per (edge, channel))
// -------------------------------------------------------------------------
__global__ void edge_aggregate_kernel(const float* __restrict__ xl,
                                      const int* __restrict__ src,
                                      const int* __restrict__ dst,
                                      const float* __restrict__ logit,
                                      const float* __restrict__ den,
                                      float* __restrict__ out,
                                      int H, int C) {
    int tid = blockIdx.x * blockDim.x + threadIdx.x;
    const int HC = H * C;
    if (tid >= ET * HC) return;
    int e = tid / HC;
    int i = tid - e * HC;
    int h = i / C;
    int s, d;
    if (e < N_EDGES) { s = src[e]; d = dst[e]; } else { s = d = e - N_EDGES; }
    float alpha = logit[(size_t)e * H + h] / (den[(size_t)d * H + h] + 1e-16f);
    atomicAdd(out + (size_t)d * HC + i, alpha * xl[(size_t)s * HC + i]);
}

// -------------------------------------------------------------------------
// bias + ELU -> next layer's f16 features          (layers 1,2)
// bias only  -> final f32 output                   (layer 3)
// -------------------------------------------------------------------------
__global__ void finalize_elu_kernel(const float* __restrict__ acc,
                                    const float* __restrict__ bias,
                                    _Float16* __restrict__ xh_next, int HC) {
    int tid = blockIdx.x * blockDim.x + threadIdx.x;
    if (tid >= N_NODES * HC) return;
    int i = tid % HC;
    float v = acc[tid] + bias[i];
    v = v > 0.0f ? v : expm1f(v);
    xh_next[tid] = (_Float16)v;
}

__global__ void finalize_out_kernel(const float* __restrict__ acc,
                                    const float* __restrict__ bias,
                                    float* __restrict__ out, int HC) {
    int tid = blockIdx.x * blockDim.x + threadIdx.x;
    if (tid >= N_NODES * HC) return;
    int i = tid % HC;
    out[tid] = acc[tid] + bias[i];
}

// -------------------------------------------------------------------------
// host side
// -------------------------------------------------------------------------
static inline int cdiv(long long a, long long b) { return (int)((a + b - 1) / b); }

extern "C" void kernel_launch(void* const* d_in, const int* in_sizes, int n_in,
                              void* d_out, int out_size, void* d_ws, size_t ws_size,
                              hipStream_t stream) {
    const float* x   = (const float*)d_in[0];
    const int*   ei  = (const int*)d_in[1];
    const int*   src = ei;
    const int*   dst = ei + N_EDGES;

    // workspace carving (256B aligned)
    char* ws = (char*)d_ws;
    size_t off = 0;
    auto carve = [&](size_t bytes) {
        char* p = ws + off;
        off += (bytes + 255) & ~(size_t)255;
        return p;
    };
    _Float16* xh    = (_Float16*)carve((size_t)N_NODES * K_DIM * 2); // f16 features
    _Float16* wpl   = (_Float16*)carve((size_t)8 * KSTEPS * 32 * 16 * 2); // packed Wl
    _Float16* wpr   = (_Float16*)carve((size_t)8 * KSTEPS * 32 * 16 * 2); // packed Wr
    float*    xl    = (float*)   carve((size_t)N_NODES * 128 * 4);
    float*    xr    = (float*)   carve((size_t)N_NODES * 128 * 4);
    float*    logit = (float*)   carve((size_t)ET * 4 * 4);
    float*    mbuf  = (float*)   carve((size_t)N_NODES * 4 * 4);
    float*    den   = (float*)   carve((size_t)N_NODES * 4 * 4);
    float*    accum = (float*)   carve((size_t)N_NODES * 128 * 4);
    (void)ws_size; (void)n_in; (void)in_sizes; (void)out_size;

    const int B = 256;

    // x (f32) -> xh (f16)
    {
        int n = N_NODES * K_DIM;
        f32_to_f16_kernel<<<cdiv(n, B), B, 0, stream>>>(x, xh, n);
    }

    auto run_layer = [&](const float* Wl, const float* Wr, const float* att,
                         const float* bias, int H, int C, bool last) {
        const int HC      = H * C;       // output width (128 or 40)
        const int tiles_n = (HC + 15) / 16;
        const int waves   = (N_NODES / 16) * tiles_n;

        // pack weights into WMMA B layout (fused f32->f16 + padding)
        int pn = tiles_n * KSTEPS * 32 * 16;
        pack_w_kernel<<<cdiv(pn, B), B, 0, stream>>>(Wl, wpl, HC, tiles_n);
        pack_w_kernel<<<cdiv(pn, B), B, 0, stream>>>(Wr, wpr, HC, tiles_n);

        // fused dual GEMM via WMMA
        int gblocks = cdiv(waves, B / 32);
        gemm2_wmma_f16_kernel<<<gblocks, B, 0, stream>>>(
            xh, wpl, wpr, xl, xr, N_NODES, HC, tiles_n);

        // init segment buffers
        fill_f32_kernel<<<cdiv(N_NODES * H, B), B, 0, stream>>>(mbuf, -INFINITY, N_NODES * H);
        fill_f32_kernel<<<cdiv(N_NODES * H, B), B, 0, stream>>>(den, 0.0f, N_NODES * H);
        fill_f32_kernel<<<cdiv(N_NODES * HC, B), B, 0, stream>>>(accum, 0.0f, N_NODES * HC);

        // segment softmax + aggregation
        edge_logit_kernel<<<cdiv((long long)ET * H, B), B, 0, stream>>>(
            xl, xr, src, dst, att, logit, mbuf, H, C);
        edge_expsum_kernel<<<cdiv((long long)ET * H, B), B, 0, stream>>>(
            dst, mbuf, logit, den, H);
        edge_aggregate_kernel<<<cdiv((long long)ET * HC, B), B, 0, stream>>>(
            xl, src, dst, logit, den, accum, H, C);

        // epilogue
        if (!last) {
            finalize_elu_kernel<<<cdiv(N_NODES * HC, B), B, 0, stream>>>(
                accum, bias, xh, HC);
        } else {
            finalize_out_kernel<<<cdiv(N_NODES * HC, B), B, 0, stream>>>(
                accum, bias, (float*)d_out, HC);
        }
    };

    run_layer((const float*)d_in[2],  (const float*)d_in[3],
              (const float*)d_in[4],  (const float*)d_in[5],  4, 32, false);
    run_layer((const float*)d_in[6],  (const float*)d_in[7],
              (const float*)d_in[8],  (const float*)d_in[9],  4, 32, false);
    run_layer((const float*)d_in[10], (const float*)d_in[11],
              (const float*)d_in[12], (const float*)d_in[13], 1, 40, true);
}
